// MATGCNBlock_9689446220046
// MI455X (gfx1250) — matrix operand order repaired and checked by reading, not compiled, not run on hardware
//
#include <hip/hip_runtime.h>
#include <hip/hip_bf16.h>

// MATGCN block for MI455X (gfx1250, wave32).
// Heavy GEMMs -> v_wmma_f32_16x16x32_bf16, LDS double-buffered B slabs staged
// with GLOBAL_LOAD_ASYNC_TO_LDS_B128 (builtin confirmed present; params are
// pointers to clang vector type 'int __attribute__((vector_size(16)))').
// Everything else: fused bandwidth-bound VALU kernels.

typedef __bf16 bf16;
typedef __attribute__((ext_vector_type(16))) __bf16 v16bf;
typedef __attribute__((ext_vector_type(8)))  __bf16 v8bf;
typedef __attribute__((ext_vector_type(8)))  float  v8f;
typedef int v4i __attribute__((vector_size(16)));  // matches builtin param type

#define DEVINL static __device__ __forceinline__

#if defined(__has_builtin)
#if __has_builtin(__builtin_amdgcn_global_load_async_to_lds_b128)
#define USE_ASYNC_LDS 1
#endif
#endif

// 16-byte global -> LDS stage (async DMA when available, register bounce else)
DEVINL void stage16(const void* src, void* dst) {
#ifdef USE_ASYNC_LDS
  __builtin_amdgcn_global_load_async_to_lds_b128((v4i*)src, (v4i*)dst, 0, 0);
#else
  *(v4i*)dst = *(const v4i*)src;
#endif
}

// wait own async copies, then block barrier (slab visible to all waves after)
DEVINL void async_fence_block() {
#ifdef USE_ASYNC_LDS
#if __has_builtin(__builtin_amdgcn_s_wait_asynccnt)
  __builtin_amdgcn_s_wait_asynccnt(0);
#else
  asm volatile("s_wait_asynccnt 0x0" ::: "memory");
#endif
#endif
  __syncthreads();
}

DEVINL float wave_sum(float v) {
#pragma unroll
  for (int o = 16; o > 0; o >>= 1) v += __shfl_down(v, o, 32);
  return v;
}

DEVINL v16bf cat8(v8bf lo, v8bf hi) {
  return __builtin_shufflevector(lo, hi, 0,1,2,3,4,5,6,7,8,9,10,11,12,13,14,15);
}

// ---------------------------------------------------------------------------
// B-operand swizzle: store B[K][Ncol] (bf16) as [K/32][N/16][lane][16] so each
// lane's WMMA B fragment is one contiguous 32B chunk.
// Lane layout this code uses for the B fragment (contiguous k, as A): lane = (n&15) | ((k>>4&1)<<4),
// element j = k&15.
// ---------------------------------------------------------------------------

// x [8][64][24000] f32 -> bf16 swizzled B for GEMM1 (K=64, N=24000)
__global__ void k_swz_x(const float* __restrict__ x, bf16* __restrict__ xsw) {
  long idx = (long)blockIdx.x * 256 + threadIdx.x;
  if (idx >= 8ll * 64 * 24000) return;
  int b = (int)(idx / (64 * 24000));
  int rem = (int)(idx - (long)b * 64 * 24000);
  int k = rem / 24000;
  int ncol = rem % 24000;
  float v = x[idx];
  int kt = k >> 5, ntt = ncol >> 4;
  int lane = (ncol & 15) | (((k >> 4) & 1) << 4);
  int elem = k & 15;
  long dst = (((long)b * 2 + kt) * 1500 + ntt) * 512 + lane * 16 + elem;
  xsw[dst] = (bf16)v;
}

// x1 [8][64][24000] f32 -> xg bf16 swizzled B for GEMM2 (K=1024 padded, N=1536)
// B[k=m][col=c*24+t] = x1[b][c][m][t]; rows k>=1000 zero-filled.
__global__ void k_swz_x1(const float* __restrict__ x1, bf16* __restrict__ xgsw) {
  long idx = (long)blockIdx.x * 256 + threadIdx.x;
  if (idx >= 8ll * 1024 * 1536) return;
  int b = (int)(idx / (1024 * 1536));
  int rem = (int)(idx - (long)b * 1024 * 1536);
  int k = rem / 1536;
  int col = rem % 1536;
  int c = col / 24, t = col % 24;
  float v = (k < 1000) ? x1[((long)(b * 64 + c)) * 24000 + k * 24 + t] : 0.0f;
  int kt = k >> 5, ctt = col >> 4;
  int lane = (col & 15) | (((k >> 4) & 1) << 4);
  int elem = k & 15;
  long dst = (((long)b * 32 + kt) * 96 + ctt) * 512 + lane * 16 + elem;
  xgsw[dst] = (bf16)v;
}

// ---------------------------------------------------------------------------
// bf16 WMMA GEMM: C[z][M][N] f32 = A[z][M][K] (bf16, row major, lda) x swizzled
// B. One wave owns a 16x64 tile (4 accumulators), 8 waves per block sharing a
// double-buffered 4KB B k-slab in LDS (async-staged). K mult of 32, N mult of
// 64. Ragged M: clamped A-row loads + guarded stores (EXEC all-ones through
// WMMA as the ISA requires).
// ---------------------------------------------------------------------------
__global__ __launch_bounds__(256)
void k_wmma_gemm(const bf16* __restrict__ A, const bf16* __restrict__ Bsw,
                 float* __restrict__ C, int M, int K, int N, int lda, int ldc,
                 long sA, long sB, long sC) {
  __shared__ __align__(16) bf16 Bs[2][2048];
  const int tid = threadIdx.x;
  const int lane = tid & 31;
  const int wave = tid >> 5;
  const int z = blockIdx.z;
  const int m0 = (blockIdx.x * 8 + wave) * 16;
  const int n0 = blockIdx.y * 64;
  const int ntiles = N >> 4;
  int mrow = m0 + (lane & 15);
  if (mrow >= M) mrow = M - 1;  // clamp: valid reads, dead rows
  const bf16* Ap = A + (long)z * sA + (long)mrow * lda + ((lane >> 4) << 4);
  // 4-tile slab for k-chunk kc starts at (kc*ntiles + n0/16)*512; each of the
  // 256 threads moves one 16B chunk (256*8 elems = 2048 = full slab).
  const bf16* Bslab = Bsw + (long)z * sB + (long)(n0 >> 4) * 512 + tid * 8;
  const long slabStride = (long)ntiles * 512;
  const bf16* bs = &Bs[0][0] + lane * 16;  // per-lane fragment base
  v8f a0 = {}, a1 = {}, a2 = {}, a3 = {};
  const int nk = K >> 5;
  stage16(Bslab, &Bs[0][tid * 8]);
  async_fence_block();
  for (int kc = 0; kc < nk; ++kc) {
    int cur = kc & 1;
    if (kc + 1 < nk)  // prefetch next slab into the other buffer
      stage16(Bslab + (kc + 1) * slabStride, &Bs[cur ^ 1][tid * 8]);
    const bf16* bc = bs + cur * 2048;
    v16bf b0 = cat8(*(const v8bf*)(bc),        *(const v8bf*)(bc + 8));
    v16bf b1 = cat8(*(const v8bf*)(bc + 512),  *(const v8bf*)(bc + 520));
    v16bf b2 = cat8(*(const v8bf*)(bc + 1024), *(const v8bf*)(bc + 1032));
    v16bf b3 = cat8(*(const v8bf*)(bc + 1536), *(const v8bf*)(bc + 1544));
    // A 16x32 fragment: lanes 0-15 k-base 0, lanes 16-31 k-base 16;
    // elems 0-15 -> kbase+j.
    v16bf av = cat8(*(const v8bf*)(Ap + kc * 32), *(const v8bf*)(Ap + kc * 32 + 8));
    a0 = __builtin_amdgcn_wmma_f32_16x16x32_bf16(false, av, false, b0, (short)0, a0, false, false);
    a1 = __builtin_amdgcn_wmma_f32_16x16x32_bf16(false, av, false, b1, (short)0, a1, false, false);
    a2 = __builtin_amdgcn_wmma_f32_16x16x32_bf16(false, av, false, b2, (short)0, a2, false, false);
    a3 = __builtin_amdgcn_wmma_f32_16x16x32_bf16(false, av, false, b3, (short)0, a3, false, false);
    async_fence_block();  // slab kc fully consumed; next slab landed
  }
  // C/D layout: VGPR v -> row m0+v (lanes 0-15) / m0+8+v (lanes 16-31).
  const int nb = n0 + (lane & 15);
  const int mh = (lane >> 4) * 8;
#pragma unroll
  for (int v = 0; v < 8; ++v) {
    int cm = m0 + mh + v;
    if (cm < M) {
      float* cp = C + (long)z * sC + (long)cm * ldc + nb;
      cp[0] = a0[v]; cp[16] = a1[v]; cp[32] = a2[v]; cp[48] = a3[v];
    }
  }
}

// ---------------------------------------------------------------------------
// att0 projections: P1[b,i,r] = sum_k xf[b,i,k] W1[k,r];
//                   P2[b,i,r] = sum_k W2[r,k] xf[b,i,k].  K = 24000.
// ---------------------------------------------------------------------------
__global__ __launch_bounds__(256)
void k_att0_P(const float* __restrict__ x, const float* __restrict__ W1,
              const float* __restrict__ W2, float* __restrict__ P1,
              float* __restrict__ P2) {
  int bi = blockIdx.x;  // b*64 + i
  const float* row = x + (long)bi * 24000;
  float a1[10] = {}, a2[10] = {};
  for (int q = threadIdx.x; q < 24000; q += 256) {
    float xv = row[q];
    const float* w1 = W1 + (long)q * 10;
#pragma unroll
    for (int r = 0; r < 10; ++r) {
      a1[r] += xv * w1[r];
      a2[r] += xv * W2[(long)r * 24000 + q];
    }
  }
  __shared__ float part[8][20];
  int lane = threadIdx.x & 31, wave = threadIdx.x >> 5;
#pragma unroll
  for (int r = 0; r < 10; ++r) {
    float s1 = wave_sum(a1[r]);
    float s2 = wave_sum(a2[r]);
    if (lane == 0) { part[wave][r] = s1; part[wave][10 + r] = s2; }
  }
  __syncthreads();
  if (threadIdx.x < 20) {
    float s = 0;
    for (int w = 0; w < 8; ++w) s += part[w][threadIdx.x];
    if (threadIdx.x < 10) P1[bi * 10 + threadIdx.x] = s;
    else                  P2[bi * 10 + threadIdx.x - 10] = s;
  }
}

// att0 softmax over 64, emits bf16 A-operand for GEMM1.
__global__ __launch_bounds__(64)
void k_att0_softmax(const float* __restrict__ P1, const float* __restrict__ P2,
                    bf16* __restrict__ A0) {
  int bi = blockIdx.x;  // b*64 + i
  int b = bi >> 6;
  int j = threadIdx.x;
  float s = 0;
#pragma unroll
  for (int r = 0; r < 10; ++r) s += P1[bi * 10 + r] * P2[(b * 64 + j) * 10 + r];
  s *= rsqrtf(24000.0f);
  __shared__ float red[64];
  __shared__ float mxs, sums;
  red[j] = s; __syncthreads();
  if (j == 0) { float m = -1e30f; for (int t = 0; t < 64; ++t) m = fmaxf(m, red[t]); mxs = m; }
  __syncthreads();
  float e = __expf(s - mxs);
  red[j] = e; __syncthreads();
  if (j == 0) { float t0 = 0; for (int t = 0; t < 64; ++t) t0 += red[t]; sums = t0; }
  __syncthreads();
  A0[bi * 64 + j] = (bf16)(e / sums);
}

// gatt projections from x1 (strided gather): K = 1536 = (c,t).
__global__ __launch_bounds__(256)
void k_gatt_P(const float* __restrict__ x1, const float* __restrict__ W1,
              const float* __restrict__ W2, float* __restrict__ P1,
              float* __restrict__ P2) {
  int bi = blockIdx.x;  // b*1000 + n
  int b = bi / 1000, n = bi % 1000;
  float a1[10] = {}, a2[10] = {};
  for (int q = threadIdx.x; q < 1536; q += 256) {
    int c = q / 24, t = q % 24;
    float xv = x1[((long)(b * 64 + c)) * 24000 + n * 24 + t];
    const float* w1 = W1 + (long)q * 10;
#pragma unroll
    for (int r = 0; r < 10; ++r) {
      a1[r] += xv * w1[r];
      a2[r] += xv * W2[(long)r * 1536 + q];
    }
  }
  __shared__ float part[8][20];
  int lane = threadIdx.x & 31, wave = threadIdx.x >> 5;
#pragma unroll
  for (int r = 0; r < 10; ++r) {
    float s1 = wave_sum(a1[r]);
    float s2 = wave_sum(a2[r]);
    if (lane == 0) { part[wave][r] = s1; part[wave][10 + r] = s2; }
  }
  __syncthreads();
  if (threadIdx.x < 20) {
    float s = 0;
    for (int w = 0; w < 8; ++w) s += part[w][threadIdx.x];
    if (threadIdx.x < 10) P1[bi * 10 + threadIdx.x] = s;
    else                  P2[bi * 10 + threadIdx.x - 10] = s;
  }
}

// gatt softmax over 1000 cols, gate by A_adj, emit bf16 A (K padded to 1024).
__global__ __launch_bounds__(256)
void k_gatt_softmax(const float* __restrict__ P1, const float* __restrict__ P2,
                    const float* __restrict__ Aadj, bf16* __restrict__ Ag) {
  int bi = blockIdx.x;  // b*1000 + i
  int b = bi / 1000, i = bi % 1000;
  float p1[10];
#pragma unroll
  for (int r = 0; r < 10; ++r) p1[r] = P1[bi * 10 + r];
  const float sc = rsqrtf(1536.0f);
  float sv[4];
  float mx = -1e30f;
#pragma unroll
  for (int jj = 0; jj < 4; ++jj) {
    int j = threadIdx.x + jj * 256;
    float s = -1e30f;
    if (j < 1000) {
      s = 0;
#pragma unroll
      for (int r = 0; r < 10; ++r) s += p1[r] * P2[(b * 1000 + j) * 10 + r];
      s *= sc;
    }
    sv[jj] = s;
    mx = fmaxf(mx, s);
  }
  __shared__ float red[8];
  __shared__ float mxs, sums;
  int lane = threadIdx.x & 31, wave = threadIdx.x >> 5;
  float m = mx;
#pragma unroll
  for (int o = 16; o > 0; o >>= 1) m = fmaxf(m, __shfl_down(m, o, 32));
  if (lane == 0) red[wave] = m;
  __syncthreads();
  if (threadIdx.x == 0) { float mm = red[0]; for (int w = 1; w < 8; ++w) mm = fmaxf(mm, red[w]); mxs = mm; }
  __syncthreads();
  float lsum = 0;
#pragma unroll
  for (int jj = 0; jj < 4; ++jj) {
    int j = threadIdx.x + jj * 256;
    if (j < 1000) { sv[jj] = __expf(sv[jj] - mxs); lsum += sv[jj]; }
  }
  float s2 = wave_sum(lsum);
  if (lane == 0) red[wave] = s2;
  __syncthreads();
  if (threadIdx.x == 0) { float t = 0; for (int w = 0; w < 8; ++w) t += red[w]; sums = t; }
  __syncthreads();
  float inv = 1.0f / sums;
#pragma unroll
  for (int jj = 0; jj < 4; ++jj) {
    int j = threadIdx.x + jj * 256;
    if (j < 1024) {
      float v = (j < 1000) ? sv[jj] * inv * Aadj[(long)i * 1000 + j] : 0.0f;
      Ag[(long)bi * 1024 + j] = (bf16)v;
    }
  }
}

// gcn channel mix g[b,o,n,t] = sum_c g1[b,n,c,t] W[c,o], fused with partial
// tatt projections (avoids a strided re-read of g).
__global__ __launch_bounds__(256)
void k_gcn_mix(const float* __restrict__ g1, const float* __restrict__ W,
               const float* __restrict__ W1t, const float* __restrict__ W2t,
               float* __restrict__ g, float* __restrict__ Ppart) {
  int bi = blockIdx.x;  // b*1000 + n
  int b = bi / 1000, n = bi % 1000;
  __shared__ __align__(16) float s1[1536];
  __shared__ float sg[1536];
  for (int q = threadIdx.x; q < 384; q += 256)  // 1536 f32 = 384 x 16B
    stage16(g1 + (long)bi * 1536 + q * 4, s1 + q * 4);
  async_fence_block();
  for (int p = threadIdx.x; p < 1536; p += 256) {
    int o = p / 24, t = p % 24;
    float acc = 0;
#pragma unroll 8
    for (int c = 0; c < 64; ++c) acc += s1[c * 24 + t] * W[c * 64 + o];
    sg[o * 24 + t] = acc;
    g[(((long)b * 64 + o) * 1000 + n) * 24 + t] = acc;
  }
  __syncthreads();
  if (threadIdx.x < 240) {
    int t = threadIdx.x / 10, r = threadIdx.x % 10;
    float p1 = 0, p2 = 0;
    for (int o = 0; o < 64; ++o) {
      float v = sg[o * 24 + t];
      p1 += v * W1t[((long)n * 64 + o) * 10 + r];
      p2 += v * W2t[(long)r * 64000 + n * 64 + o];
    }
    Ppart[(long)bi * 480 + threadIdx.x] = p1;
    Ppart[(long)bi * 480 + 240 + threadIdx.x] = p2;
  }
}

// reduce Ppart over the 1000 graph nodes -> Pt[8][480]
__global__ void k_reduce_Pt(const float* __restrict__ Ppart, float* __restrict__ Pt) {
  int idx = blockIdx.x * 256 + threadIdx.x;  // 0..3839
  int b = idx / 480, j = idx % 480;
  float s = 0;
  for (int n = 0; n < 1000; ++n) s += Ppart[((long)b * 1000 + n) * 480 + j];
  Pt[idx] = s;
}

// temporal attention softmax (24x24 per batch)
__global__ void k_tatt_softmax(const float* __restrict__ Pt, float* __restrict__ At) {
  int b = blockIdx.x, i = threadIdx.x;
  if (i >= 24) return;
  float p1[10];
#pragma unroll
  for (int r = 0; r < 10; ++r) p1[r] = Pt[b * 480 + i * 10 + r];
  const float sc = rsqrtf(64000.0f);
  float s[24];
  float mx = -1e30f;
  for (int j = 0; j < 24; ++j) {
    float v = 0;
#pragma unroll
    for (int r = 0; r < 10; ++r) v += p1[r] * Pt[b * 480 + 240 + j * 10 + r];
    v *= sc;
    s[j] = v;
    mx = fmaxf(mx, v);
  }
  float sum = 0;
  for (int j = 0; j < 24; ++j) { s[j] = __expf(s[j] - mx); sum += s[j]; }
  float inv = 1.0f / sum;
  for (int j = 0; j < 24; ++j) At[(b * 24 + i) * 24 + j] = s[j] * inv;
}

// Fused tail: tatt apply + dilated causal convs + residual + ReLU + LayerNorm.
// One block per (b,n); all (Co x T) tiles live in LDS; slab loads are async.
__global__ __launch_bounds__(256)
void k_final(const float* __restrict__ g, const float* __restrict__ At,
             const float* __restrict__ x, const float* __restrict__ w1,
             const float* __restrict__ b1, const float* __restrict__ w2,
             const float* __restrict__ b2, const float* __restrict__ rw,
             const float* __restrict__ rb, const float* __restrict__ lng,
             const float* __restrict__ lnb, float* __restrict__ out) {
  int bi = blockIdx.x;  // b*1000 + n
  int b = bi / 1000, n = bi % 1000;
  __shared__ __align__(16) float at[576];
  __shared__ __align__(16) float bufA[1536], bufB[1536], bufC[1536];
  __shared__ float mu[24], rs[24];
  for (int q = threadIdx.x; q < 144; q += 256)  // 576 f32 = 144 x 16B
    stage16(At + b * 576 + q * 4, at + q * 4);
  for (int q = threadIdx.x; q < 384; q += 256) {  // g slab: 64 rows x 96B
    int o = q / 6, j = q - o * 6;
    stage16(g + ((((long)b * 64 + o) * 1000 + n) * 24 + j * 4), bufA + q * 4);
  }
  async_fence_block();
  // x2[o,t] = sum_t' At[t,t'] * g[o,t']
  for (int p = threadIdx.x; p < 1536; p += 256) {
    int o = p / 24, t = p % 24;
    float a = 0;
#pragma unroll
    for (int tp = 0; tp < 24; ++tp) a += at[t * 24 + tp] * bufA[o * 24 + tp];
    bufB[p] = a;
  }
  __syncthreads();
  // conv1, dilation 1, causal
  for (int p = threadIdx.x; p < 1536; p += 256) {
    int o = p / 24, t = p % 24;
    float a = b1[o];
    const float* w = w1 + o * 128;
    for (int i2 = 0; i2 < 64; ++i2) {
      float xp = (t >= 1) ? bufB[i2 * 24 + t - 1] : 0.0f;
      a += w[i2 * 2] * xp + w[i2 * 2 + 1] * bufB[i2 * 24 + t];
    }
    bufC[p] = fmaxf(a, 0.0f);
  }
  __syncthreads();
  // conv2, dilation 2, causal
  for (int p = threadIdx.x; p < 1536; p += 256) {
    int o = p / 24, t = p % 24;
    float a = b2[o];
    const float* w = w2 + o * 128;
    for (int i2 = 0; i2 < 64; ++i2) {
      float xp = (t >= 2) ? bufC[i2 * 24 + t - 2] : 0.0f;
      a += w[i2 * 2] * xp + w[i2 * 2 + 1] * bufC[i2 * 24 + t];
    }
    bufA[p] = fmaxf(a, 0.0f);
  }
  __syncthreads();
  // original input slab for the 1x1 residual conv
  for (int q = threadIdx.x; q < 384; q += 256) {
    int c = q / 6, j = q - c * 6;
    stage16(x + ((long)(b * 64 + c)) * 24000 + n * 24 + j * 4, bufB + q * 4);
  }
  async_fence_block();
  for (int p = threadIdx.x; p < 1536; p += 256) {
    int o = p / 24, t = p % 24;
    float a = rb[o];
    const float* w = rw + o * 64;
#pragma unroll 8
    for (int c = 0; c < 64; ++c) a += w[c] * bufB[c * 24 + t];
    bufC[p] = fmaxf(bufA[p] + a, 0.0f);
  }
  __syncthreads();
  if (threadIdx.x < 24) {
    int t = threadIdx.x;
    float m = 0;
    for (int o = 0; o < 64; ++o) m += bufC[o * 24 + t];
    m *= (1.0f / 64.0f);
    float v = 0;
    for (int o = 0; o < 64; ++o) { float d = bufC[o * 24 + t] - m; v += d * d; }
    v *= (1.0f / 64.0f);
    mu[t] = m;
    rs[t] = rsqrtf(v + 1e-5f);
  }
  __syncthreads();
  for (int p = threadIdx.x; p < 1536; p += 256) {
    int o = p / 24, t = p % 24;
    out[(((long)b * 64 + o) * 1000 + n) * 24 + t] =
        (bufC[p] - mu[t]) * rs[t] * lng[o] + lnb[o];
  }
}

// ---------------------------------------------------------------------------
extern "C" void kernel_launch(void* const* d_in, const int* in_sizes, int n_in,
                              void* d_out, int out_size, void* d_ws, size_t ws_size,
                              hipStream_t stream) {
  (void)in_sizes; (void)n_in; (void)out_size; (void)ws_size;
  const float* x      = (const float*)d_in[0];
  const float* A_adj  = (const float*)d_in[1];
  const float* att0W1 = (const float*)d_in[2];
  const float* att0W2 = (const float*)d_in[3];
  const float* gattW1 = (const float*)d_in[4];
  const float* gattW2 = (const float*)d_in[5];
  const float* gcnW   = (const float*)d_in[6];
  const float* tattW1 = (const float*)d_in[7];
  const float* tattW2 = (const float*)d_in[8];
  const float* c1w = (const float*)d_in[9];
  const float* c1b = (const float*)d_in[10];
  const float* c2w = (const float*)d_in[11];
  const float* c2b = (const float*)d_in[12];
  const float* rw  = (const float*)d_in[13];
  const float* rb  = (const float*)d_in[14];
  const float* lng = (const float*)d_in[15];
  const float* lnb = (const float*)d_in[16];
  float* out = (float*)d_out;

  char* ws = (char*)d_ws;
  size_t cur = 0;
  auto alloc = [&](size_t bytes) -> void* {
    void* p = ws + cur;
    cur += (bytes + 255) & ~size_t(255);
    return p;
  };
  bf16*  xbf  = (bf16*)alloc(8ull * 2 * 1500 * 512 * 2);   // swizzled B, GEMM1
  bf16*  a0bf = (bf16*)alloc(8ull * 64 * 64 * 2);          // A operand, GEMM1
  float* P1_0 = (float*)alloc(512ull * 10 * 4);
  float* P2_0 = (float*)alloc(512ull * 10 * 4);
  float* x1   = (float*)alloc(8ull * 64 * 24000 * 4);      // x1, later reused as g1
  bf16*  xgsw = (bf16*)alloc(8ull * 32 * 96 * 512 * 2);    // swizzled B, GEMM2
  float* P1g  = (float*)alloc(8000ull * 10 * 4);
  float* P2g  = (float*)alloc(8000ull * 10 * 4);
  bf16*  agbf = (bf16*)alloc(8ull * 1000 * 1024 * 2);      // A operand, GEMM2
  float* gbuf = (float*)alloc(8ull * 64 * 24000 * 4);
  float* Ppart= (float*)alloc(8ull * 1000 * 480 * 4);
  float* Pt   = (float*)alloc(8ull * 480 * 4);
  float* At   = (float*)alloc(8ull * 576 * 4);

  k_swz_x<<<48000, 256, 0, stream>>>(x, xbf);
  k_att0_P<<<512, 256, 0, stream>>>(x, att0W1, att0W2, P1_0, P2_0);
  k_att0_softmax<<<512, 64, 0, stream>>>(P1_0, P2_0, a0bf);
  // GEMM1: x1 = A0 @ xf   (M=64, K=64, N=24000) per batch
  k_wmma_gemm<<<dim3(1, 375, 8), 256, 0, stream>>>(
      a0bf, xbf, x1, 64, 64, 24000, 64, 24000,
      (long)64 * 64, (long)2 * 1500 * 512, (long)64 * 24000);
  k_swz_x1<<<49152, 256, 0, stream>>>(x1, xgsw);
  k_gatt_P<<<8000, 256, 0, stream>>>(x1, gattW1, gattW2, P1g, P2g);
  k_gatt_softmax<<<8000, 256, 0, stream>>>(P1g, P2g, A_adj, agbf);
  // GEMM2: g1 = Ag @ xg   (M=1000, K=1024, N=1536) per batch; reuses x1 buffer
  k_wmma_gemm<<<dim3(8, 24, 8), 256, 0, stream>>>(
      agbf, xgsw, x1, 1000, 1024, 1536, 1024, 1536,
      (long)1000 * 1024, (long)32 * 96 * 512, (long)1000 * 1536);
  k_gcn_mix<<<8000, 256, 0, stream>>>(x1, gcnW, tattW1, tattW2, gbuf, Ppart);
  k_reduce_Pt<<<15, 256, 0, stream>>>(Ppart, Pt);
  k_tatt_softmax<<<8, 32, 0, stream>>>(Pt, At);
  k_final<<<8000, 256, 0, stream>>>(gbuf, At, x, c1w, c1b, c2w, c2b,
                                    rw, rb, lng, lnb, out);
}
